// Quantize_topk_38362647888303
// MI455X (gfx1250) — matrix operand order repaired and checked by reading, not compiled, not run on hardware
//
#include <hip/hip_runtime.h>

typedef __attribute__((ext_vector_type(16))) _Float16 v16h;
typedef __attribute__((ext_vector_type(8)))  float    v8f;

#define NROWS   16384          // 16*32*32
#define DIMS    256
#define NCODES  2048
#define KSEL    4
#define KSTEPS  (DIMS / 32)    // 8 wmma k-steps
#define MTILES  (NROWS / 16)   // 1024
#define NTILES  (NCODES / 16)  // 128

// ---- workspace layout (bytes) ----
#define PA_OFF  0
#define PA_SIZE ((size_t)MTILES * KSTEPS * 32 * 16 * 2)   // 8 MiB  A fragments (f16)
#define PB_OFF  (PA_OFF + PA_SIZE)
#define PB_SIZE ((size_t)NTILES * KSTEPS * 32 * 16 * 2)   // 1 MiB  B fragments (f16)
#define ET_OFF  (PB_OFF + PB_SIZE)
#define ET_SIZE ((size_t)NCODES * DIMS * 4)               // 2 MiB  codebook.T (f32)
#define BN_OFF  (ET_OFF + ET_SIZE)
#define BN_SIZE ((size_t)NCODES * 4)                      // 8 KiB  0.5*||e||^2
#define TK_OFF  (BN_OFF + BN_SIZE)                        // 256 KiB top-4 indices

#define NEG_INF (-3.402823466e38f)

// A fragment (16x32 f16): lane l (M = l&15); half e: K = s*32 + (l>=16 ? 16 : 0) + e
__global__ __launch_bounds__(256) void pack_a(const float* __restrict__ in,
                                              _Float16* __restrict__ pA) {
    int mt = blockIdx.x;
    int tid = threadIdx.x;
    int s = tid >> 5, l = tid & 31;
    int row = mt * 16 + (l & 15);
    int kbase = s * 32 + ((l & 16) ? 16 : 0);
    const float* src = in + (size_t)row * DIMS;
    v16h frag;
#pragma unroll
    for (int e = 0; e < 16; e++) {
        int k = kbase + e;
        frag[e] = (_Float16)src[k];
    }
    *((v16h*)pA + (size_t)(mt * KSTEPS + s) * 32 + l) = frag;
}

// B fragment (32x16 f16): lane l (N = l&15); half e: K = s*32 + (l>=16 ? 16 : 0) + e
__global__ __launch_bounds__(256) void pack_b(const float* __restrict__ emb,
                                              _Float16* __restrict__ pB) {
    int t = blockIdx.x;
    int tid = threadIdx.x;
    int s = tid >> 5, l = tid & 31;
    int col = t * 16 + (l & 15);
    int kbase = s * 32 + ((l & 16) ? 16 : 0);
    v16h frag;
#pragma unroll
    for (int e = 0; e < 16; e++)
        frag[e] = (_Float16)emb[(size_t)(kbase + e) * NCODES + col];
    *((v16h*)pB + (size_t)(t * KSTEPS + s) * 32 + l) = frag;
}

// Per-code half squared norm + transposed codebook; also zero the diff slot each call.
__global__ __launch_bounds__(256) void norm_transpose(const float* __restrict__ emb,
                                                      float* __restrict__ embT,
                                                      float* __restrict__ bnorm,
                                                      float* __restrict__ diff_slot) {
    int j = blockIdx.x;      // code 0..2047
    int d = threadIdx.x;     // dim 0..255
    float v = emb[(size_t)d * NCODES + j];
    embT[(size_t)j * DIMS + d] = v;
    __shared__ float red[256];
    red[d] = v * v;
    __syncthreads();
    for (int o = 128; o > 0; o >>= 1) {
        if (d < o) red[d] += red[d + o];
        __syncthreads();
    }
    if (d == 0) bnorm[j] = 0.5f * red[0];
    if (j == 0 && d == 0) *diff_slot = 0.0f;
}

// One block = 16 rows x 2048 codes. 8 waves; wave w owns column tiles w, w+8, ...
__global__ __launch_bounds__(256) void gemm_topk(const _Float16* __restrict__ pA,
                                                 const _Float16* __restrict__ pB,
                                                 const float* __restrict__ bnorm,
                                                 int* __restrict__ topk) {
    __shared__ float sc[16 * NCODES];      // 128 KiB score buffer
    __shared__ float cs[16][16][KSEL];     // stage-1 candidate scores
    __shared__ int   ci[16][16][KSEL];     // stage-1 candidate indices
    int mt  = blockIdx.x;
    int tid = threadIdx.x;
    int w = tid >> 5;
    int l = tid & 31;

    // A fragments resident in VGPRs (64 VGPRs; wave32 budget is up to 1024)
    const v16h* Aex = (const v16h*)pA + (size_t)mt * KSTEPS * 32;
    v16h a[KSTEPS];
#pragma unroll
    for (int s = 0; s < KSTEPS; s++) a[s] = Aex[s * 32 + l];

    const v16h* Bex = (const v16h*)pB;
    for (int ti = 0; ti < NTILES / 8; ti++) {
        int t = ti * 8 + w;
        if (t + 8 < NTILES)  // prefetch next B tile -> global_prefetch_b8
            __builtin_prefetch((const void*)(Bex + (size_t)(t + 8) * KSTEPS * 32 + l), 0, 1);
        v8f acc = {};
#pragma unroll
        for (int s = 0; s < KSTEPS; s++) {
            v16h b = Bex[(size_t)(t * KSTEPS + s) * 32 + l];
            acc = __builtin_amdgcn_wmma_f32_16x16x32_f16(
                false, a[s], false, b, (short)0, acc, false, false);
        }
        int col = t * 16 + (l & 15);
        float bn = bnorm[col];
        int mhalf = (l & 16) ? 8 : 0;   // C/D layout: lanes 16-31 hold M=8..15
#pragma unroll
        for (int r = 0; r < 8; r++)
            sc[(mhalf + r) * NCODES + col] = acc[r] - bn;
    }
    __syncthreads();

    // stage 1: thread scans 128 columns of one row, keeps local top-4
    {
        int m = tid >> 4;   // row 0..15
        int p = tid & 15;   // segment 0..15
        const float* rowp = &sc[m * NCODES + p * 128];
        float s0 = NEG_INF, s1 = NEG_INF, s2 = NEG_INF, s3 = NEG_INF;
        int   i0 = 0, i1 = 0, i2 = 0, i3 = 0;
        for (int c = 0; c < 128; c++) {
            float v = rowp[c];
            int idx = p * 128 + c;
            if (v > s3) {
                if (v > s0)      { s3=s2;i3=i2; s2=s1;i2=i1; s1=s0;i1=i0; s0=v;i0=idx; }
                else if (v > s1) { s3=s2;i3=i2; s2=s1;i2=i1; s1=v;i1=idx; }
                else if (v > s2) { s3=s2;i3=i2; s2=v;i2=idx; }
                else             { s3=v;i3=idx; }
            }
        }
        cs[m][p][0]=s0; cs[m][p][1]=s1; cs[m][p][2]=s2; cs[m][p][3]=s3;
        ci[m][p][0]=i0; ci[m][p][1]=i1; ci[m][p][2]=i2; ci[m][p][3]=i3;
    }
    __syncthreads();

    // stage 2: 16 threads merge 16x4 candidates per row (ascending p keeps
    // smallest-index-first tie-breaking, matching jax argmax/top_k)
    if (tid < 16) {
        int m = tid;
        float s0 = NEG_INF, s1 = NEG_INF, s2 = NEG_INF, s3 = NEG_INF;
        int   i0 = 0, i1 = 0, i2 = 0, i3 = 0;
        for (int p = 0; p < 16; p++) {
#pragma unroll
            for (int k = 0; k < KSEL; k++) {
                float v = cs[m][p][k];
                int idx = ci[m][p][k];
                if (v > s3) {
                    if (v > s0)      { s3=s2;i3=i2; s2=s1;i2=i1; s1=s0;i1=i0; s0=v;i0=idx; }
                    else if (v > s1) { s3=s2;i3=i2; s2=s1;i2=i1; s1=v;i1=idx; }
                    else if (v > s2) { s3=s2;i3=i2; s2=v;i2=idx; }
                    else             { s3=v;i3=idx; }
                }
            }
        }
        int row = mt * 16 + m;
        topk[row * KSEL + 0] = i0;
        topk[row * KSEL + 1] = i1;
        topk[row * KSEL + 2] = i2;
        topk[row * KSEL + 3] = i3;
    }
}

// One block per row: gather 4 code rows, straight-through output, MSE partial.
__global__ __launch_bounds__(256) void gather_out(const float* __restrict__ input,
                                                  const float* __restrict__ embT,
                                                  const int* __restrict__ topk,
                                                  float* __restrict__ out) {
    int n = blockIdx.x;
    int d = threadIdx.x;
    float* out_topk = out;                                   // [N, K*D]
    float* diff     = out + (size_t)NROWS * KSEL * DIMS;     // scalar
    float* out_st   = diff + 1;                              // [N, D]

    int i0 = topk[n * KSEL + 0];
    int i1 = topk[n * KSEL + 1];
    int i2 = topk[n * KSEL + 2];
    int i3 = topk[n * KSEL + 3];

    float q0 = embT[(size_t)i0 * DIMS + d];
    size_t obase = (size_t)n * (KSEL * DIMS) + d;
    out_topk[obase + 0 * DIMS] = q0;
    out_topk[obase + 1 * DIMS] = embT[(size_t)i1 * DIMS + d];
    out_topk[obase + 2 * DIMS] = embT[(size_t)i2 * DIMS + d];
    out_topk[obase + 3 * DIMS] = embT[(size_t)i3 * DIMS + d];
    out_st[(size_t)n * DIMS + d] = q0;    // x + stopgrad(q - x) == q

    float e = q0 - input[(size_t)n * DIMS + d];
    __shared__ float red[256];
    red[d] = e * e;
    __syncthreads();
    for (int o = 128; o > 0; o >>= 1) {
        if (d < o) red[d] += red[d + o];
        __syncthreads();
    }
    if (d == 0)
        atomicAdd(diff, red[0] * (1.0f / ((float)NROWS * (float)DIMS)));
}

extern "C" void kernel_launch(void* const* d_in, const int* in_sizes, int n_in,
                              void* d_out, int out_size, void* d_ws, size_t ws_size,
                              hipStream_t stream) {
    const float* inp = (const float*)d_in[0];   // [16,32,32,256] f32
    const float* emb = (const float*)d_in[1];   // [256,2048] f32
    float* out = (float*)d_out;

    char* ws = (char*)d_ws;
    _Float16* pA   = (_Float16*)(ws + PA_OFF);
    _Float16* pB   = (_Float16*)(ws + PB_OFF);
    float*    embT = (float*)(ws + ET_OFF);
    float*    bn   = (float*)(ws + BN_OFF);
    int*      tk   = (int*)(ws + TK_OFF);
    float*    diff = out + (size_t)NROWS * KSEL * DIMS;

    pack_a<<<MTILES, 256, 0, stream>>>(inp, pA);
    pack_b<<<NTILES, 256, 0, stream>>>(emb, pB);
    norm_transpose<<<NCODES, 256, 0, stream>>>(emb, embT, bn, diff);
    gemm_topk<<<MTILES, 256, 0, stream>>>(pA, pB, bn, tk);
    gather_out<<<NROWS, 256, 0, stream>>>(inp, embT, tk, out);
}
